// Block_22986664968285
// MI455X (gfx1250) — matrix operand order repaired and checked by reading, not compiled, not run on hardware
//
#include <hip/hip_runtime.h>
#include <stdint.h>

#define DEV __device__ __forceinline__
#define ASYNC_LDS 1
#if __has_builtin(__builtin_amdgcn_tensor_load_to_lds)
#define HAS_TDM 1
#else
#define HAS_TDM 0
#endif

typedef __attribute__((ext_vector_type(16))) __bf16    v16bf;
typedef __attribute__((ext_vector_type(8)))  float     v8f;
typedef __attribute__((ext_vector_type(4)))  unsigned  v4u;
typedef __attribute__((ext_vector_type(8)))  int       v8i;
typedef __attribute__((ext_vector_type(4)))  int       v4i;

union FragU { v16bf v; uint4 q[2]; };

// f32 -> bf16 single (compiler lowers to v_cvt_pk_bf16_f32)
DEV unsigned short f2bf(float a) {
  return __builtin_bit_cast(unsigned short, (__bf16)a);
}
// f32 x2 -> packed bf16x2 in ONE v_cvt_pk_bf16_f32 (mnemonic disasm-confirmed)
DEV unsigned pack_bf16(float a, float b) {
  unsigned r;
  asm("v_cvt_pk_bf16_f32 %0, %1, %2" : "=v"(r) : "v"(a), "v"(b));
  return r;
}

DEV float tanh_fast(float x) {
#if __has_builtin(__builtin_amdgcn_tanh_f32)
  return __builtin_amdgcn_tanh_f32(x);
#elif __has_builtin(__builtin_amdgcn_tanhf)
  return __builtin_amdgcn_tanhf(x);
#else
  return tanhf(x);
#endif
}

DEV float exp2_fast(float x) {
#if __has_builtin(__builtin_amdgcn_exp2f)
  return __builtin_amdgcn_exp2f(x);    // raw v_exp_f32 (2^x)
#else
  return exp2f(x);
#endif
}

DEV v8f wmma_bf16(v16bf a, v16bf b, v8f c) {
  return __builtin_amdgcn_wmma_f32_16x16x32_bf16(false, a, false, b, (short)0, c, false, false);
}

// ---------------------------------------------------------------------------
// Weight transpose + f32 -> bf16 convert:  in[K,N] f32  ->  out[N,K] bf16
// ---------------------------------------------------------------------------
__global__ void transpose_cvt_kernel(const float* __restrict__ in,
                                     unsigned short* __restrict__ out,
                                     int K, int N) {
  __shared__ float tile[32][33];
  const int tx = threadIdx.x, ty = threadIdx.y;   // block 32x8
  const int n0 = blockIdx.x * 32, k0 = blockIdx.y * 32;
  #pragma unroll
  for (int i = 0; i < 32; i += 8)
    tile[ty + i][tx] = in[(size_t)(k0 + ty + i) * N + (n0 + tx)];
  __syncthreads();
  #pragma unroll
  for (int i = 0; i < 32; i += 8)
    out[(size_t)(n0 + ty + i) * K + (k0 + tx)] = f2bf(tile[tx][ty + i]);
}

// ---------------------------------------------------------------------------
// Row LayerNorm over D=1024, f32 in -> bf16 out. One wave per row, float4 IO.
// ---------------------------------------------------------------------------
__global__ __launch_bounds__(256, 1) void ln_row_kernel(
    const float* __restrict__ x, const float* __restrict__ g,
    const float* __restrict__ b, unsigned short* __restrict__ out, int rows) {
  const int lane = threadIdx.x & 31;
  const int wave = threadIdx.x >> 5;
  const int row  = blockIdx.x * 8 + wave;
  if (row >= rows) return;
  const float4* xr4 = (const float4*)(x + (size_t)row * 1024);
  const float4* g4p = (const float4*)g;
  const float4* b4p = (const float4*)b;
  float4 v[8];
  float s = 0.f, sq = 0.f;
  #pragma unroll
  for (int t = 0; t < 8; t++) {
    float4 f = xr4[lane + 32 * t];
    v[t] = f;
    s  += f.x + f.y + f.z + f.w;
    sq += f.x * f.x + f.y * f.y + f.z * f.z + f.w * f.w;
  }
  #pragma unroll
  for (int m = 1; m < 32; m <<= 1) {
    s  += __shfl_xor(s,  m, 32);
    sq += __shfl_xor(sq, m, 32);
  }
  const float mean = s * (1.f / 1024.f);
  const float var  = sq * (1.f / 1024.f) - mean * mean;
  const float rs   = rsqrtf(var + 1e-6f);
  uint2* orow = (uint2*)(out + (size_t)row * 1024);
  #pragma unroll
  for (int t = 0; t < 8; t++) {
    const int c = lane + 32 * t;
    float4 gg = g4p[c], bb = b4p[c];
    uint2 pk;
    pk.x = pack_bf16((v[t].x - mean) * rs * gg.x + bb.x,
                     (v[t].y - mean) * rs * gg.y + bb.y);
    pk.y = pack_bf16((v[t].z - mean) * rs * gg.z + bb.z,
                     (v[t].w - mean) * rs * gg.w + bb.w);
    orow[c] = pk;
  }
}

// ---------------------------------------------------------------------------
// GEMM: C[M,N] = A[M,K](bf16) x Bt[N,K](bf16), f32 accum via WMMA.
// Block = 256 thr (8 waves); wave tile 32(M) x 128(N); block tile 256x128.
// B tile (128x32 bf16 = 8KB) double-buffer staged in LDS:
//   preferred: one TDM tensor_load_to_lds descriptor per k-step (TENSORcnt),
//   fallback:  per-lane global_load_async_to_lds_b128 (ASYNCcnt).
// EPI 0: store f32.  EPI 1: out = res + (acc+bias)*ls (f32).
// EPI 2: out = bf16(gelu_tanh(acc+bias)).
// ---------------------------------------------------------------------------
template <int EPI>
__global__ __launch_bounds__(256, 1) void gemm_bf16_kernel(
    const unsigned short* __restrict__ A, const unsigned short* __restrict__ Bt,
    int M, int N, int K,
    float* __restrict__ outF, unsigned short* __restrict__ outH,
    const float* __restrict__ bias, const float* __restrict__ ls,
    const float* __restrict__ res) {
  __shared__ __align__(16) unsigned short bsh[2][128 * 32];   // 2 x 8KB
  const int tid  = threadIdx.x;
  const int lane = tid & 31;
  const int wave = tid >> 5;
  const int half = lane >> 4;
  const int l15  = lane & 15;
  const int wm   = blockIdx.y * 256 + wave * 32;
  const int ncol = blockIdx.x * 128;

  const unsigned ldsTile[2] = {(unsigned)(uintptr_t)&bsh[0][0],
                               (unsigned)(uintptr_t)&bsh[1][0]};
  // per-lane staging geometry (async/sync fallback): 512 x 16B pieces
  const int p0r = tid >> 2,         p0c = (tid & 3) * 8;
  const int p1r = (tid + 256) >> 2, p1c = ((tid + 256) & 3) * 8;
  const unsigned short* bsrc0 = Bt + (size_t)(ncol + p0r) * K + p0c;
  const unsigned short* bsrc1 = Bt + (size_t)(ncol + p1r) * K + p1c;
  const unsigned ldsP0[2] = {(unsigned)(uintptr_t)&bsh[0][p0r * 32 + p0c],
                             (unsigned)(uintptr_t)&bsh[1][p0r * 32 + p0c]};
  const unsigned ldsP1[2] = {(unsigned)(uintptr_t)&bsh[0][p1r * 32 + p1c],
                             (unsigned)(uintptr_t)&bsh[1][p1r * 32 + p1c]};

  auto stage = [&](int buf, int k0) {
#if HAS_TDM
    if (wave == 0) {
      // D# per CDNA5 ISA: 2D tile 32(K) x 128(N) bf16
      unsigned long long ga =
          (unsigned long long)(uintptr_t)(Bt + (size_t)ncol * K + k0);
      v4u g0 = {1u,                                   // count=1, user mode
                ldsTile[buf],                         // lds_addr
                (unsigned)ga,                         // global_addr[31:0]
                (unsigned)((ga >> 32) & 0x1ffffffu) | (2u << 30)};  // addr[56:32], type=2
      v8i g1 = {(int)(1u << 16),                      // wg_mask=0, data_size=1 (2B)
                (int)(((unsigned)K & 0xffffu) << 16), // tensor_dim0 [15:0]
                (int)((((unsigned)K >> 16) & 0xffffu) | (0x4000u << 16)), // dim0 hi, dim1 lo
                (int)(32u << 16),                     // dim1 hi=0, tile_dim0=32
                (int)128u,                            // tile_dim1=128, tile_dim2=0
                (int)K,                               // tensor_dim0_stride lo32
                0, 0};                                // stride hi, stride1=0
      __builtin_amdgcn_tensor_load_to_lds(g0, g1, (v4i){0, 0, 0, 0},
                                          (v4i){0, 0, 0, 0},
                                          (v8i){0, 0, 0, 0, 0, 0, 0, 0}, 0);
    }
#elif ASYNC_LDS
    unsigned long long g0 = (unsigned long long)(uintptr_t)(bsrc0 + k0);
    unsigned long long g1 = (unsigned long long)(uintptr_t)(bsrc1 + k0);
    asm volatile("global_load_async_to_lds_b128 %0, %1, off"
                 :: "v"(ldsP0[buf]), "v"(g0) : "memory");
    asm volatile("global_load_async_to_lds_b128 %0, %1, off"
                 :: "v"(ldsP1[buf]), "v"(g1) : "memory");
#else
    *(uint4*)&bsh[buf][p0r * 32 + p0c] = *(const uint4*)(bsrc0 + k0);
    *(uint4*)&bsh[buf][p1r * 32 + p1c] = *(const uint4*)(bsrc1 + k0);
#endif
  };
  auto stage_wait = [&]() {
#if HAS_TDM
    if (wave == 0) __builtin_amdgcn_s_wait_tensorcnt(0);
#elif ASYNC_LDS
    asm volatile("s_wait_asynccnt 0" ::: "memory");
#endif
    __syncthreads();
  };

  const unsigned short* arow0 = A + (size_t)(wm + l15) * K;
  const unsigned short* arow1 = arow0 + (size_t)16 * K;

  v8f acc[2][8];
  #pragma unroll
  for (int mf = 0; mf < 2; mf++)
    #pragma unroll
    for (int nt = 0; nt < 8; nt++) acc[mf][nt] = (v8f){};

  stage(0, 0);
  stage_wait();
  int buf = 0;
  for (int k0 = 0; k0 < K; k0 += 32) {
    const bool more = (k0 + 32) < K;
    if (more) stage(buf ^ 1, k0 + 32);
    FragU a0, a1;
    const int ka = k0 + half * 8;           // A lane K-chunks {ka, 16+ka}
    a0.q[0] = *(const uint4*)(arow0 + ka);
    a0.q[1] = *(const uint4*)(arow0 + ka + 16);
    a1.q[0] = *(const uint4*)(arow1 + ka);
    a1.q[1] = *(const uint4*)(arow1 + ka + 16);
    const unsigned short* bb = &bsh[buf][0];
    #pragma unroll
    for (int nt = 0; nt < 8; nt++) {
      FragU bm;                              // B lane: K-chunks {half*8, 16+half*8}
      const unsigned short* bp = bb + (nt * 16 + l15) * 32 + half * 8;
      bm.q[0] = ((const uint4*)bp)[0];
      bm.q[1] = ((const uint4*)(bp + 16))[0];
      acc[0][nt] = wmma_bf16(a0.v, bm.v, acc[0][nt]);
      acc[1][nt] = wmma_bf16(a1.v, bm.v, acc[1][nt]);
    }
    if (more) stage_wait();
    buf ^= 1;
  }

  #pragma unroll
  for (int nt = 0; nt < 8; nt++) {
    const int n = ncol + nt * 16 + l15;
    float bi = 0.f, lv = 0.f;
    if constexpr (EPI != 0) bi = bias[n];
    if constexpr (EPI == 1) lv = ls[n];
    #pragma unroll
    for (int mf = 0; mf < 2; mf++) {
      #pragma unroll
      for (int i = 0; i < 8; i++) {
        const int m = wm + mf * 16 + i + half * 8;
        float v = acc[mf][nt][i];
        if constexpr (EPI == 0) {
          outF[(size_t)m * N + n] = v;
        } else if constexpr (EPI == 1) {
          outF[(size_t)m * N + n] = res[(size_t)m * N + n] + (v + bi) * lv;
        } else {
          v += bi;
          float gl = 0.5f * v * (1.f + tanh_fast(0.7978845608f * (v + 0.044715f * v * v * v)));
          outH[(size_t)m * N + n] = f2bf(gl);
        }
      }
    }
  }
}

// ---------------------------------------------------------------------------
// QKV post-processing: per-head QK-LayerNorm -> bf16; q additionally scaled by
// HD^-0.5 * log2(e) so attention can use raw exp2 (v_exp_f32) for softmax.
// V -> bf16 transposed [bh, d, p]. One wave per (b,h,p); 2 dims per lane.
// ---------------------------------------------------------------------------
DEV float2 ln64(float x0, float x1, float g0, float g1, float b0, float b1) {
  float s = x0 + x1, sq = x0 * x0 + x1 * x1;
  for (int m = 1; m < 32; m <<= 1) {
    s  += __shfl_xor(s,  m, 32);
    sq += __shfl_xor(sq, m, 32);
  }
  const float mean = s * (1.f / 64.f);
  const float var  = sq * (1.f / 64.f) - mean * mean;
  const float rs   = rsqrtf(var + 1e-6f);
  return make_float2((x0 - mean) * rs * g0 + b0, (x1 - mean) * rs * g1 + b1);
}

__global__ __launch_bounds__(256, 1) void qkv_post_kernel(
    const float* __restrict__ qkv,
    const float* __restrict__ qg, const float* __restrict__ qbb,
    const float* __restrict__ kg, const float* __restrict__ kbb,
    unsigned short* __restrict__ qo, unsigned short* __restrict__ ko,
    unsigned short* __restrict__ vt) {
  const int lane = threadIdx.x & 31;
  const int wave = threadIdx.x >> 5;
  const int gid  = blockIdx.x * 8 + wave;   // over B*H*P = 73728
  const int p  = gid % 576;
  const int bh = gid / 576;
  const int d0 = lane * 2;
  const int h = bh & 15, b = bh >> 4;
  const float* base = qkv + (size_t)(b * 576 + p) * 3072 + h * 64 + d0;
  const size_t rowo = (size_t)(bh * 576 + p) * 64;
  const float QS = 0.125f * 1.44269504088896340736f;   // HD^-0.5 * log2(e)
  {
    float2 q = *(const float2*)(base);
    float2 r = ln64(q.x, q.y, qg[d0], qg[d0 + 1], qbb[d0], qbb[d0 + 1]);
    *(unsigned*)(qo + rowo + d0) = pack_bf16(r.x * QS, r.y * QS);
  }
  {
    float2 k = *(const float2*)(base + 1024);
    float2 r = ln64(k.x, k.y, kg[d0], kg[d0 + 1], kbb[d0], kbb[d0 + 1]);
    *(unsigned*)(ko + rowo + d0) = pack_bf16(r.x, r.y);
  }
  {
    float2 v = *(const float2*)(base + 2048);
    vt[((size_t)bh * 64 + d0)     * 576 + p] = f2bf(v.x);
    vt[((size_t)bh * 64 + d0 + 1) * 576 + p] = f2bf(v.y);
  }
}

// ---------------------------------------------------------------------------
// Attention. grid = (P/64, B*H), block = 128 (4 waves, fully independent).
// QK-LayerNorm bounds |score| <= ~8*log2e, so softmax needs no max
// subtraction; log2(e) is folded into q so p = exp2(score) directly.
// Key interleave: S0 scores EVEN keys (kt+2j), S1 ODD keys (kt+2j+1), so a
// lane's (p0,p1) pair sits at adjacent K-positions 2j,2j+1 in the A layout
// -> one packed-cvt + one u32 ds_store per row, conflict-free; the
// position<->key mapping stays identity so the V fragment is unchanged.
// Per-lane partial row sums reduced once after the key loop. P transposed
// C->A layout through a wave-private LDS tile (s_wait_dscnt 0 fence).
// ---------------------------------------------------------------------------
__global__ __launch_bounds__(128, 1) void attn_kernel(
    const unsigned short* __restrict__ qb, const unsigned short* __restrict__ kb,
    const unsigned short* __restrict__ vt, float* __restrict__ outp) {
  __shared__ __align__(16) unsigned short pshm[4][16 * 32];
  const int lane = threadIdx.x & 31;
  const int wave = threadIdx.x >> 5;
  const int half = lane >> 4;
  const int l15  = lane & 15;
  const int bh   = blockIdx.y;
  const int h = bh & 15, b = bh >> 4;
  const int q0 = blockIdx.x * 64 + wave * 16;

  const unsigned short* qbase = qb + (size_t)bh * 576 * 64;
  const unsigned short* kbase = kb + (size_t)bh * 576 * 64;
  const unsigned short* vbase = vt + (size_t)bh * 64 * 576;

  FragU aq0, aq1;
  {
    const unsigned short* qr = qbase + (size_t)(q0 + l15) * 64;
    const int kbs = half * 16;
    aq0.q[0] = *(const uint4*)(qr + kbs);
    aq0.q[1] = *(const uint4*)(qr + 8 + kbs);
    aq1.q[0] = *(const uint4*)(qr + 32 + kbs);
    aq1.q[1] = *(const uint4*)(qr + 40 + kbs);
  }

  v8f oc[4];
  #pragma unroll
  for (int i = 0; i < 4; i++) oc[i] = (v8f){};
  float lsum[8];
  #pragma unroll
  for (int i = 0; i < 8; i++) lsum[i] = 0.f;
  unsigned short* myp = &pshm[wave][0];
  unsigned* myp32 = (unsigned*)myp;

  for (int kt = 0; kt < 576; kt += 32) {
    v8f s0 = (v8f){}, s1 = (v8f){};
    {  // even keys: kt + 2j
      const unsigned short* kr = kbase + (size_t)(kt + 2 * l15) * 64;
      FragU b0, b1;
      b0.q[0] = ((const uint4*)(kr + half * 16))[0];
      b0.q[1] = ((const uint4*)(kr + half * 16))[1];
      b1.q[0] = ((const uint4*)(kr + 32 + half * 16))[0];
      b1.q[1] = ((const uint4*)(kr + 32 + half * 16))[1];
      s0 = wmma_bf16(aq0.v, b0.v, s0);
      s0 = wmma_bf16(aq1.v, b1.v, s0);
    }
    {  // odd keys: kt + 2j + 1
      const unsigned short* kr = kbase + (size_t)(kt + 2 * l15 + 1) * 64;
      FragU b0, b1;
      b0.q[0] = ((const uint4*)(kr + half * 16))[0];
      b0.q[1] = ((const uint4*)(kr + half * 16))[1];
      b1.q[0] = ((const uint4*)(kr + 32 + half * 16))[0];
      b1.q[1] = ((const uint4*)(kr + 32 + half * 16))[1];
      s1 = wmma_bf16(aq0.v, b0.v, s1);
      s1 = wmma_bf16(aq1.v, b1.v, s1);
    }
    // exp2 + partial row sums; one packed cvt + one u32 store per row
    #pragma unroll
    for (int i = 0; i < 8; i++) {
      const float p0 = exp2_fast(s0[i]);
      const float p1 = exp2_fast(s1[i]);
      lsum[i] += p0 + p1;
      myp32[(i + half * 8) * 16 + l15] = pack_bf16(p0, p1);
    }
    asm volatile("s_wait_dscnt 0" ::: "memory");   // wave-private LDS fence
    FragU ap;  // P as A fragment (16x32)
    {
      const unsigned short* pr = myp + l15 * 32 + half * 16;
      ap.q[0] = *(const uint4*)(pr);
      ap.q[1] = *(const uint4*)(pr + 8);
    }
    #pragma unroll
    for (int dt = 0; dt < 4; dt++) {
      FragU bv;
      const unsigned short* vr =
          vbase + (size_t)(dt * 16 + l15) * 576 + kt + half * 16;
      bv.q[0] = ((const uint4*)vr)[0];
      bv.q[1] = ((const uint4*)vr)[1];
      oc[dt] = wmma_bf16(ap.v, bv.v, oc[dt]);
    }
  }

  #pragma unroll
  for (int i = 0; i < 8; i++) {
    float rs = lsum[i];
    rs += __shfl_xor(rs, 1, 32);
    rs += __shfl_xor(rs, 2, 32);
    rs += __shfl_xor(rs, 4, 32);
    rs += __shfl_xor(rs, 8, 32);
    const float inv = 1.f / rs;
    const int p = q0 + i + half * 8;
    float* orow = outp + (size_t)(b * 576 + p) * 1024 + h * 64;
    #pragma unroll
    for (int dt = 0; dt < 4; dt++) orow[dt * 16 + l15] = oc[dt][i] * inv;
  }
}

// ---------------------------------------------------------------------------
extern "C" void kernel_launch(void* const* d_in, const int* in_sizes, int n_in,
                              void* d_out, int out_size, void* d_ws, size_t ws_size,
                              hipStream_t stream) {
  const float* x      = (const float*)d_in[0];
  const float* w_qkv  = (const float*)d_in[1];
  const float* q_g    = (const float*)d_in[2];
  const float* q_b    = (const float*)d_in[3];
  const float* k_g    = (const float*)d_in[4];
  const float* k_b    = (const float*)d_in[5];
  const float* o_g    = (const float*)d_in[6];
  const float* o_b    = (const float*)d_in[7];
  const float* w_proj = (const float*)d_in[8];
  const float* b_proj = (const float*)d_in[9];
  const float* ln1_g  = (const float*)d_in[10];
  const float* ln1_b  = (const float*)d_in[11];
  const float* ln2_g  = (const float*)d_in[12];
  const float* ln2_b  = (const float*)d_in[13];
  const float* w1     = (const float*)d_in[14];
  const float* b1     = (const float*)d_in[15];
  const float* w2     = (const float*)d_in[16];
  const float* b2     = (const float*)d_in[17];
  const float* ls1    = (const float*)d_in[18];
  const float* ls2    = (const float*)d_in[19];

  char* ws = (char*)d_ws;
  size_t off = 0;
  auto alloc = [&](size_t bytes) -> void* {
    void* p = ws + off;
    off = (off + bytes + 255) & ~(size_t)255;
    return p;
  };

  const int BP = 8 * 576;       // 4608
  unsigned short* wqkvT = (unsigned short*)alloc((size_t)3072 * 1024 * 2);
  unsigned short* wprjT = (unsigned short*)alloc((size_t)1024 * 1024 * 2);
  unsigned short* w1T   = (unsigned short*)alloc((size_t)4096 * 1024 * 2);
  unsigned short* w2T   = (unsigned short*)alloc((size_t)1024 * 4096 * 2);
  unsigned short* xn    = (unsigned short*)alloc((size_t)BP * 1024 * 2);
  float*          qkvF  = (float*)alloc((size_t)BP * 3072 * 4);
  unsigned short* qB    = (unsigned short*)alloc((size_t)128 * 576 * 64 * 2);
  unsigned short* kB    = (unsigned short*)alloc((size_t)128 * 576 * 64 * 2);
  unsigned short* vT    = (unsigned short*)alloc((size_t)128 * 576 * 64 * 2);
  float*          attnF = (float*)alloc((size_t)BP * 1024 * 4);
  float*          x1    = (float*)alloc((size_t)BP * 1024 * 4);
  unsigned short* hB    = (unsigned short*)alloc((size_t)BP * 4096 * 2);

  const dim3 tb(32, 8);
  transpose_cvt_kernel<<<dim3(3072 / 32, 1024 / 32), tb, 0, stream>>>(w_qkv, wqkvT, 1024, 3072);
  transpose_cvt_kernel<<<dim3(1024 / 32, 1024 / 32), tb, 0, stream>>>(w_proj, wprjT, 1024, 1024);
  transpose_cvt_kernel<<<dim3(4096 / 32, 1024 / 32), tb, 0, stream>>>(w1, w1T, 1024, 4096);
  transpose_cvt_kernel<<<dim3(1024 / 32, 4096 / 32), tb, 0, stream>>>(w2, w2T, 4096, 1024);

  // attention branch
  ln_row_kernel<<<BP / 8, 256, 0, stream>>>(x, ln1_g, ln1_b, xn, BP);
  gemm_bf16_kernel<0><<<dim3(3072 / 128, BP / 256), 256, 0, stream>>>(
      xn, wqkvT, BP, 3072, 1024, qkvF, nullptr, nullptr, nullptr, nullptr);
  qkv_post_kernel<<<(128 * 576) / 8, 256, 0, stream>>>(qkvF, q_g, q_b, k_g, k_b, qB, kB, vT);
  attn_kernel<<<dim3(576 / 64, 128), 128, 0, stream>>>(qB, kB, vT, attnF);
  ln_row_kernel<<<BP / 8, 256, 0, stream>>>(attnF, o_g, o_b, xn, BP);
  gemm_bf16_kernel<1><<<dim3(1024 / 128, BP / 256), 256, 0, stream>>>(
      xn, wprjT, BP, 1024, 1024, x1, nullptr, b_proj, ls1, x);

  // MLP branch
  ln_row_kernel<<<BP / 8, 256, 0, stream>>>(x1, ln2_g, ln2_b, xn, BP);
  gemm_bf16_kernel<2><<<dim3(4096 / 128, BP / 256), 256, 0, stream>>>(
      xn, w1T, BP, 4096, 1024, nullptr, hB, b1, nullptr, nullptr);
  gemm_bf16_kernel<1><<<dim3(1024 / 128, BP / 256), 256, 0, stream>>>(
      hB, w2T, BP, 1024, 4096, (float*)d_out, nullptr, b2, ls2, x1);

  (void)in_sizes; (void)n_in; (void)out_size; (void)ws_size;
}
